// OpensetFastRCNNOutputLayers_18090402250919
// MI455X (gfx1250) — matrix operand order repaired and checked by reading, not compiled, not run on hardware
//
#include <hip/hip_runtime.h>

typedef _Float16 v16h __attribute__((ext_vector_type(16)));
typedef _Float16 v8h  __attribute__((ext_vector_type(8)));
typedef _Float16 v4h  __attribute__((ext_vector_type(4)));
typedef float    v8f  __attribute__((ext_vector_type(8)));

#define DDIM 1024
#define NCOL_OUT 320
#define NTILES 21          // 21 * 16 = 336 padded columns (320 bbox + 1 iou + 15 zero)
#define ROWS_PER_BLOCK 128
#define KC 32              // K per chunk == WMMA K
#define NCHUNK (DDIM / KC)
#define XS 40              // padded half-stride of X tile rows (80B, conflict-free)
#define WS 40              // padded half-stride of W^T tile rows (80B, conflict-free)

__device__ __forceinline__ v16h ld_frag32(const _Float16* p) {
    v8h a = *(const v8h*)p;
    v8h b = *(const v8h*)(p + 8);
    return __builtin_shufflevector(a, b, 0,1,2,3,4,5,6,7,8,9,10,11,12,13,14,15);
}

__global__ __launch_bounds__(256)
void rcnn_head_wmma(const float* __restrict__ x,
                    const float* __restrict__ Wb,
                    const float* __restrict__ bb,
                    const float* __restrict__ Wi,
                    const float* __restrict__ bi,
                    float* __restrict__ out,
                    int Nrows)
{
    __shared__ _Float16 ldsX[2][ROWS_PER_BLOCK * XS];   // 2 x 10240 B
    __shared__ _Float16 ldsWt[2][NTILES * 16 * WS];     // 2 x 26880 B (transposed: [n][k])

    const int tid  = threadIdx.x;
    const int lane = tid & 31;
    const int wid  = tid >> 5;
    const int lo   = lane & 15;
    const int hi   = lane >> 4;
    const int rowBase = blockIdx.x * ROWS_PER_BLOCK;

    v8f acc[NTILES];
    {
        v8f z = {0.f, 0.f, 0.f, 0.f, 0.f, 0.f, 0.f, 0.f};
        #pragma unroll
        for (int t = 0; t < NTILES; ++t) acc[t] = z;
    }

    // ---- staging helper: chunk at kBase -> LDS buffer `buf` (f32 -> f16) ----
    auto stage = [&](int kBase, int buf) {
        #pragma unroll
        for (int i = 0; i < 4; ++i) {
            int f   = tid + i * 256;          // 1024 float4 groups (128 rows x 8)
            int row = f >> 3;
            int k4  = f & 7;
            int g   = rowBase + row;
            if (g > Nrows - 1) g = Nrows - 1; // clamp tail rows (stores predicated later)
            const float4 v = *(const float4*)(x + (size_t)g * DDIM + kBase + k4 * 4);
            v4h h = { (_Float16)v.x, (_Float16)v.y, (_Float16)v.z, (_Float16)v.w };
            *(v4h*)(&ldsX[buf][row * XS + k4 * 4]) = h;
            if (kBase + KC < DDIM)
                __builtin_prefetch(x + (size_t)g * DDIM + kBase + KC + k4 * 4, 0, 1);
        }
        #pragma unroll
        for (int i = 0; i < 10; ++i) {
            int f  = tid + i * 256;           // 2560 float4 groups (32 k x 80)
            int k  = f / 80;
            int n4 = f - k * 80;
            const float4 v = *(const float4*)(Wb + (size_t)(kBase + k) * NCOL_OUT + n4 * 4);
            int n = n4 * 4;
            ldsWt[buf][(n + 0) * WS + k] = (_Float16)v.x;
            ldsWt[buf][(n + 1) * WS + k] = (_Float16)v.y;
            ldsWt[buf][(n + 2) * WS + k] = (_Float16)v.z;
            ldsWt[buf][(n + 3) * WS + k] = (_Float16)v.w;
        }
        if (tid < KC) ldsWt[buf][320 * WS + tid] = (_Float16)Wi[kBase + tid];  // iou col
    };

    // Zero-pad columns 321..335 once, in both buffers (never overwritten).
    for (int f = tid; f < 2 * 15 * KC; f += 256) {
        int b  = (f >= 15 * KC) ? 1 : 0;
        int f2 = f - b * 15 * KC;
        ldsWt[b][(321 + (f2 >> 5)) * WS + (f2 & 31)] = (_Float16)0.0f;
    }

    stage(0, 0);
    __syncthreads();

    // ---- main pipeline: one barrier per chunk, double-buffered LDS ----
    for (int c = 0; c < NCHUNK; ++c) {
        const int buf = c & 1;

        // A fragment: K = hi*16..+15, M = lo
        const _Float16* xrow = &ldsX[buf][(wid * 16 + lo) * XS + hi * 16];
        v8h a0 = *(const v8h*)(xrow);
        v8h a1 = *(const v8h*)(xrow + 8);
        v16h afrag = __builtin_shufflevector(a0, a1, 0,1,2,3,4,5,6,7,8,9,10,11,12,13,14,15);

        // B fragments software-pipelined one tile ahead: K = hi*16..+15, N = lo
        const _Float16* wbase = &ldsWt[buf][lo * WS + hi * 16];
        v16h bcur  = ld_frag32(wbase);
        v16h bnext = bcur;
        #pragma unroll
        for (int t = 0; t < NTILES; ++t) {
            if (t + 1 < NTILES)
                bnext = ld_frag32(wbase + (t + 1) * 16 * WS);
            acc[t] = __builtin_amdgcn_wmma_f32_16x16x32_f16(
                false, afrag, false, bcur, (short)0, acc[t], false, false);
            bcur = bnext;
        }

        if (c + 1 < NCHUNK)
            stage((c + 1) * KC, buf ^ 1);   // writes the *other* buffer
        __syncthreads();
    }

    // ---- epilogue: C/D layout is VGPR r -> (M = r + hi*8, N = lo) ----
    const int rowTileBase = rowBase + wid * 16 + hi * 8;

    #pragma unroll
    for (int t = 0; t < 20; ++t) {
        float bias = bb[t * 16 + lo];
        #pragma unroll
        for (int r = 0; r < 8; ++r) {
            int row = rowTileBase + r;
            if (row < Nrows)
                out[(size_t)row * NCOL_OUT + t * 16 + lo] = acc[t][r] + bias;
        }
    }
    {
        float biou = bi[0];
        #pragma unroll
        for (int r = 0; r < 8; ++r) {
            int row = rowTileBase + r;
            if (lo == 0 && row < Nrows) {
                float v = acc[20][r] + biou;
                out[(size_t)Nrows * NCOL_OUT + row] = 1.0f / (1.0f + __expf(-v));
            }
        }
    }
}

extern "C" void kernel_launch(void* const* d_in, const int* in_sizes, int n_in,
                              void* d_out, int out_size, void* d_ws, size_t ws_size,
                              hipStream_t stream) {
    const float* x  = (const float*)d_in[0];
    const float* Wb = (const float*)d_in[1];
    const float* bb = (const float*)d_in[2];
    const float* Wi = (const float*)d_in[3];
    const float* bi = (const float*)d_in[4];
    float* out = (float*)d_out;

    int Nrows = in_sizes[0] / DDIM;
    int grid  = (Nrows + ROWS_PER_BLOCK - 1) / ROWS_PER_BLOCK;
    rcnn_head_wmma<<<grid, 256, 0, stream>>>(x, Wb, bb, Wi, bi, out, Nrows);
}
